// GCNSAGEDiscriminator_11914239279199
// MI455X (gfx1250) — matrix operand order repaired and checked by reading, not compiled, not run on hardware
//
#include <hip/hip_runtime.h>

#define NJ      17
#define HID     256
#define OUTD    64
#define BATCH   8192
#define NODES   (BATCH * NJ)          // 139264 = 256 * 544

typedef __attribute__((ext_vector_type(16))) __bf16 v16bf;
typedef __attribute__((ext_vector_type(8)))  float  v8f;

union Frag { uint4 u[2]; v16bf v; };

// H3.6M skeleton adjacency (with self-loops), degree per joint
__device__ __constant__ int c_nbr[17][5] = {
    {0,1,4,7,0}, {1,0,2,0,0}, {2,1,3,0,0}, {3,2,0,0,0},
    {4,0,5,0,0}, {5,4,6,0,0}, {6,5,0,0,0},
    {7,0,8,0,0}, {8,7,9,11,14}, {9,8,10,0,0}, {10,9,0,0,0},
    {11,8,12,0,0}, {12,11,13,0,0}, {13,12,0,0,0},
    {14,8,15,0,0}, {15,14,16,0,0}, {16,15,0,0,0}};
__device__ __constant__ int c_deg[17] = {4,3,3,2,3,3,2,3,5,3,2,3,3,2,3,3,2};

__device__ __forceinline__ unsigned short f2bf(float f) {
    unsigned u = __float_as_uint(f);
    unsigned r = u + 0x7fffu + ((u >> 16) & 1u);   // round-to-nearest-even
    return (unsigned short)(r >> 16);
}
__device__ __forceinline__ float bf2f(unsigned short h) {
    return __uint_as_float(((unsigned)h) << 16);
}

// ---------------------------------------------------------------------------
// Build concatenated, transposed bf16 weights: Wt[n][k] with k<256 -> Wl[k][n],
// k>=256 -> Wr[k-256][n].  8 hidden convs of [256][512] + final [64][512].
// ---------------------------------------------------------------------------
__global__ __launch_bounds__(256)
void prep_weights(const float* __restrict__ Wl, const float* __restrict__ Wr,
                  const float* __restrict__ Wlf, const float* __restrict__ Wrf,
                  unsigned short* __restrict__ WT, unsigned short* __restrict__ WTF)
{
    const int idx = blockIdx.x * 256 + threadIdx.x;
    const int NH  = 8 * HID * 512;                 // 1048576
    if (idx < NH) {
        const int s   = idx >> 17;                 // conv index (l*2+i)
        const int rem = idx & 131071;
        const int n   = rem >> 9;
        const int k   = rem & 511;
        const float* W = (k < HID) ? Wl : Wr;
        const int kk   = k & 255;
        WT[idx] = f2bf(W[((size_t)s * HID + kk) * HID + n]);
    } else if (idx < NH + OUTD * 512) {
        const int r = idx - NH;
        const int n = r >> 9;
        const int k = r & 511;
        const int kk = k & 255;
        const float v = (k < HID) ? Wlf[(size_t)kk * OUTD + n]
                                  : Wrf[(size_t)kk * OUTD + n];
        WTF[r] = f2bf(v);
    }
}

// ---------------------------------------------------------------------------
// Input projection: x = kp @ Win + bin  (K=3, trivial) -> bf16
// ---------------------------------------------------------------------------
__global__ __launch_bounds__(256)
void input_layer(const float* __restrict__ kp, const float* __restrict__ Win,
                 const float* __restrict__ bin, unsigned short* __restrict__ X0)
{
    const size_t node = blockIdx.x;
    const int f = threadIdx.x;
    const float a = kp[node * 3 + 0];
    const float b = kp[node * 3 + 1];
    const float c = kp[node * 3 + 2];
    const float v = a * Win[f] + b * Win[HID + f] + c * Win[2 * HID + f] + bin[f];
    X0[node * HID + f] = f2bf(v);
}

// ---------------------------------------------------------------------------
// Mean aggregation over the fixed skeleton. One wave per node; each lane owns
// 8 bf16 features (one B128 per neighbor row).
// ---------------------------------------------------------------------------
__global__ __launch_bounds__(256)
void aggregate(const unsigned short* __restrict__ X, unsigned short* __restrict__ AGG)
{
    const int node = blockIdx.x * 8 + (threadIdx.x >> 5);
    const int lane = threadIdx.x & 31;
    const int j    = node % NJ;
    const int base = node - j;
    const int deg  = c_deg[j];
    float acc[8] = {0.f,0.f,0.f,0.f,0.f,0.f,0.f,0.f};
    for (int e = 0; e < deg; ++e) {
        const uint4 q = *(const uint4*)(X + (size_t)(base + c_nbr[j][e]) * HID + lane * 8);
        acc[0] += __uint_as_float(q.x << 16);  acc[1] += __uint_as_float(q.x & 0xffff0000u);
        acc[2] += __uint_as_float(q.y << 16);  acc[3] += __uint_as_float(q.y & 0xffff0000u);
        acc[4] += __uint_as_float(q.z << 16);  acc[5] += __uint_as_float(q.z & 0xffff0000u);
        acc[6] += __uint_as_float(q.w << 16);  acc[7] += __uint_as_float(q.w & 0xffff0000u);
    }
    const float s = 1.0f / (float)deg;
    uint4 o;
    o.x = (unsigned)f2bf(acc[0]*s) | ((unsigned)f2bf(acc[1]*s) << 16);
    o.y = (unsigned)f2bf(acc[2]*s) | ((unsigned)f2bf(acc[3]*s) << 16);
    o.z = (unsigned)f2bf(acc[4]*s) | ((unsigned)f2bf(acc[5]*s) << 16);
    o.w = (unsigned)f2bf(acc[6]*s) | ((unsigned)f2bf(acc[7]*s) << 16);
    *(uint4*)(AGG + (size_t)node * HID + lane * 8) = o;
}

// ---------------------------------------------------------------------------
// Fused dual GEMM:  Out = [Agg | Xin] (K=512) @ Wt^T + bias  (+res) (+relu)
// WG = 256 thr = 8 waves; tile M=256 (32 rows/wave), N=64. Weight panel (64KB)
// staged in LDS via CDNA5 async global->LDS; WMMA bf16 16x16x32, f32 accum.
// Each B fragment (from LDS) is reused by two WMMAs (two M-tiles).
//   mode 0: relu(out)      mode 1: relu(out + Res)      mode 2: plain (final)
// ---------------------------------------------------------------------------
__global__ __launch_bounds__(256)
void gemm_dual(const unsigned short* __restrict__ Agg,
               const unsigned short* __restrict__ Xin,
               const unsigned short* __restrict__ WT,   // [Ntot][512] bf16
               const float* __restrict__ bias,          // [Ntot]
               unsigned short* __restrict__ Out,        // [NODES][Nout]
               const unsigned short* __restrict__ Res,  // residual (mode 1)
               int mode, int Nout)
{
    __shared__ __align__(128) unsigned char ldsW[65536];   // 64 rows x 1024B

    const int tid  = threadIdx.x;
    const int nblk = blockIdx.y;

    // --- stage 64KB contiguous weight panel into LDS (async tensor path) ---
    {
        const unsigned char* src = (const unsigned char*)(WT + (size_t)nblk * 64 * 512);
        const unsigned ldsBase = (unsigned)(size_t)(&ldsW[0]);   // low 32b = LDS offset
        for (int i = tid; i < 4096; i += 256) {
            unsigned long long g = (unsigned long long)(size_t)(src + (size_t)i * 16);
            unsigned l = ldsBase + (unsigned)i * 16u;
            asm volatile("global_load_async_to_lds_b128 %0, %1, off"
                         :: "v"(l), "v"(g) : "memory");
        }
        asm volatile("s_wait_asynccnt 0" ::: "memory");
    }
    __syncthreads();

    const int lane = tid & 31;
    const int wave = tid >> 5;
    const int m    = lane & 15;
    const int half = lane >> 4;
    // two 16-row M-tiles per wave
    const size_t row0 = (size_t)blockIdx.x * 256 + (size_t)wave * 32 + m;
    const size_t row1 = row0 + 16;

    const unsigned char* aggRow0 = (const unsigned char*)(Agg + row0 * HID);
    const unsigned char* aggRow1 = (const unsigned char*)(Agg + row1 * HID);
    const unsigned char* xRow0   = (const unsigned char*)(Xin + row0 * HID);
    const unsigned char* xRow1   = (const unsigned char*)(Xin + row1 * HID);

    v8f acc[8] = {};
    #pragma unroll
    for (int ks = 0; ks < 16; ++ks) {
        const unsigned char* A0 = (ks < 8) ? aggRow0 : xRow0;
        const unsigned char* A1 = (ks < 8) ? aggRow1 : xRow1;
        const int k0 = (ks & 7) * 32;
        // A fragment (16x32 bf16): lanes 0-15: K=k0..k0+7 in V0-3,
        // K=k0+16..k0+23 in V4-7; lanes 16-31: the +8 chunks.
        Frag a0, a1;
        a0.u[0] = *(const uint4*)(A0 + (size_t)(k0 + half * 8) * 2);
        a0.u[1] = *(const uint4*)(A0 + (size_t)(k0 + half * 8) * 2 + 32);
        a1.u[0] = *(const uint4*)(A1 + (size_t)(k0 + half * 8) * 2);
        a1.u[1] = *(const uint4*)(A1 + (size_t)(k0 + half * 8) * 2 + 32);
        const int kb = ks * 32;
        #pragma unroll
        for (int j = 0; j < 4; ++j) {
            // B fragment (32x16 bf16): lane n (0-15) = column j*16+n holds
            // K=kb..kb+7 in V0-3, K=kb+16..kb+23 in V4-7; lanes 16-31 the +8 chunks.
            Frag b;
            const unsigned char* bp = ldsW + (size_t)(j * 16 + m) * 1024
                                           + (size_t)(kb + half * 8) * 2;
            b.u[0] = *(const uint4*)(bp);
            b.u[1] = *(const uint4*)(bp + 32);
            acc[j] = __builtin_amdgcn_wmma_f32_16x16x32_bf16(
                         false, a0.v, false, b.v, (short)0, acc[j], false, false);
            acc[4 + j] = __builtin_amdgcn_wmma_f32_16x16x32_bf16(
                         false, a1.v, false, b.v, (short)0, acc[4 + j], false, false);
        }
    }

    // --- epilogue: C/D layout -> lane gives N=lane&15, M = (lane>>4)*8 + r ---
    const int mrowBase = blockIdx.x * 256 + wave * 32 + half * 8;
    #pragma unroll
    for (int t = 0; t < 2; ++t) {
        #pragma unroll
        for (int j = 0; j < 4; ++j) {
            const int ncol = nblk * 64 + j * 16 + m;
            const float bv = bias[ncol];
            #pragma unroll
            for (int r = 0; r < 8; ++r) {
                const size_t orow = (size_t)(mrowBase + t * 16 + r);
                float v = acc[t * 4 + j][r] + bv;
                if (mode == 1) v += bf2f(Res[orow * HID + ncol]);
                if (mode != 2) v = fmaxf(v, 0.0f);
                Out[orow * (size_t)Nout + ncol] = f2bf(v);
            }
        }
    }
}

// ---------------------------------------------------------------------------
// Header: out[b] = g[b] . Wh + bh  over 17*64 = 1088 features. Wave per sample.
// ---------------------------------------------------------------------------
__global__ __launch_bounds__(256)
void header_k(const unsigned short* __restrict__ F, const float* __restrict__ Wh,
              const float* __restrict__ bh, float* __restrict__ out)
{
    const int b    = blockIdx.x * 8 + (threadIdx.x >> 5);
    const int lane = threadIdx.x & 31;
    const unsigned short* g = F + (size_t)b * (NJ * OUTD);
    float s = 0.f;
    for (int i = lane; i < NJ * OUTD; i += 32) s += bf2f(g[i]) * Wh[i];
    #pragma unroll
    for (int o = 16; o > 0; o >>= 1) s += __shfl_xor(s, o, 32);
    if (lane == 0) out[b] = s + bh[0];
}

// ---------------------------------------------------------------------------
extern "C" void kernel_launch(void* const* d_in, const int* in_sizes, int n_in,
                              void* d_out, int out_size, void* d_ws, size_t ws_size,
                              hipStream_t stream)
{
    (void)in_sizes; (void)n_in; (void)out_size; (void)ws_size;
    const float* kp  = (const float*)d_in[0];
    const float* Win = (const float*)d_in[1];
    const float* bin = (const float*)d_in[2];
    const float* Wl  = (const float*)d_in[3];
    const float* bl  = (const float*)d_in[4];
    const float* Wr  = (const float*)d_in[5];
    const float* Wlf = (const float*)d_in[6];
    const float* blf = (const float*)d_in[7];
    const float* Wrf = (const float*)d_in[8];
    const float* Wh  = (const float*)d_in[9];
    const float* bh  = (const float*)d_in[10];
    float* out = (float*)d_out;

    unsigned char* ws = (unsigned char*)d_ws;
    const size_t SZ = (size_t)NODES * HID * 2;               // 71.3 MB per buffer
    unsigned short* X0  = (unsigned short*)(ws);
    unsigned short* X1  = (unsigned short*)(ws + SZ);
    unsigned short* AGG = (unsigned short*)(ws + 2 * SZ);
    unsigned short* WT  = (unsigned short*)(ws + 3 * SZ);    // 8 x 256KB
    unsigned short* WTF = WT + (size_t)8 * HID * 512;        // 64KB

    prep_weights<<<4224, 256, 0, stream>>>(Wl, Wr, Wlf, Wrf, WT, WTF);
    input_layer<<<NODES, 256, 0, stream>>>(kp, Win, bin, X0);

    for (int l = 0; l < 4; ++l) {
        // h = relu(sage1(x))
        aggregate<<<NODES / 8, 256, 0, stream>>>(X0, AGG);
        gemm_dual<<<dim3(NODES / 256, 4), 256, 0, stream>>>(
            AGG, X0, WT + (size_t)(l * 2 + 0) * HID * 512,
            bl + (l * 2 + 0) * HID, X1, nullptr, 0, HID);
        // x = relu(x + sage2(h))
        aggregate<<<NODES / 8, 256, 0, stream>>>(X1, AGG);
        gemm_dual<<<dim3(NODES / 256, 4), 256, 0, stream>>>(
            AGG, X1, WT + (size_t)(l * 2 + 1) * HID * 512,
            bl + (l * 2 + 1) * HID, X0, X0, 1, HID);
    }

    // final conv -> [NODES, 64] into X1, then header
    aggregate<<<NODES / 8, 256, 0, stream>>>(X0, AGG);
    gemm_dual<<<dim3(NODES / 256, 1), 256, 0, stream>>>(
        AGG, X0, WTF, blf, X1, nullptr, 2, OUTD);
    header_k<<<BATCH / 8, 256, 0, stream>>>(X1, Wh, bh, out);
}
